// Adaptive_freqfilter_regression_4707284156638
// MI455X (gfx1250) — matrix operand order repaired and checked by reading, not compiled, not run on hardware
//
#include <hip/hip_runtime.h>
#include <hip/hip_bf16.h>
#include <math.h>

typedef __attribute__((ext_vector_type(16))) _Float16 v16h;
typedef __attribute__((ext_vector_type(8)))  float    v8f;

#define LOW_N  6291456   // 8*3*512*512
#define FQ_N   2097152   // 8*512*512
#define IMGN   262144    // 512*512

// ---------------------------------------------------------------------------
// 512-point radix-2 complex FFT in LDS. 256 threads, t = threadIdx.x.
// SIGN = -1 forward, +1 inverse (unnormalized).
// ---------------------------------------------------------------------------
template<int SIGN>
__device__ __forceinline__ void fft512(float* re, float* im, int t) {
    // bit-reversal permutation (pairs are disjoint; min-index owner swaps)
    for (int i = t; i < 512; i += 256) {
        int j = (int)(__brev((unsigned)i) >> 23);
        if (i < j) {
            float a = re[i]; re[i] = re[j]; re[j] = a;
            float b = im[i]; im[i] = im[j]; im[j] = b;
        }
    }
    __syncthreads();
    for (int s = 0; s < 9; ++s) {
        int halfm = 1 << s;
        int k   = t & (halfm - 1);
        int grp = t >> s;
        int i1 = (grp << (s + 1)) + k;
        int i2 = i1 + halfm;
        float ang = (float)SIGN * 3.14159265358979323846f * (float)k / (float)halfm;
        float sn, c;
        __sincosf(ang, &sn, &c);
        float xr = re[i2], xi = im[i2];
        float tr = c * xr - sn * xi;
        float ti = sn * xr + c * xi;
        float ur = re[i1], ui = im[i1];
        re[i2] = ur - tr; im[i2] = ui - ti;
        re[i1] = ur + tr; im[i1] = ui + ti;
        __syncthreads();
    }
}

// forward FFT along w of real input x -> complex cb
__global__ __launch_bounds__(256)
void fft_rows_real(const float* __restrict__ x, float2* __restrict__ cb) {
    __shared__ float re[512], im[512];
    int row = blockIdx.x;          // (b*3 + c)*512 + h
    int t = threadIdx.x;
    const float* src = x + (size_t)row * 512;
    for (int i = t; i < 512; i += 256) { re[i] = src[i]; im[i] = 0.0f; }
    __syncthreads();
    fft512<-1>(re, im, t);
    float2* dst = cb + (size_t)row * 512;
    for (int i = t; i < 512; i += 256) dst[i] = make_float2(re[i], im[i]);
}

// in-place complex FFT along w
template<int SIGN>
__global__ __launch_bounds__(256)
void fft_rows_c(float2* __restrict__ cb) {
    __shared__ float re[512], im[512];
    int row = blockIdx.x;
    int t = threadIdx.x;
    float2* p = cb + (size_t)row * 512;
    for (int i = t; i < 512; i += 256) { float2 z = p[i]; re[i] = z.x; im[i] = z.y; }
    __syncthreads();
    fft512<SIGN>(re, im, t);
    __syncthreads();
    for (int i = t; i < 512; i += 256) p[i] = make_float2(re[i], im[i]);
}

// in-place complex FFT along h (stride 512)
template<int SIGN>
__global__ __launch_bounds__(256)
void fft_cols_c(float2* __restrict__ cb) {
    __shared__ float re[512], im[512];
    int col = blockIdx.x;          // (b*3 + c)*512 + w
    int img = col >> 9;
    int w   = col & 511;
    int t = threadIdx.x;
    float2* base = cb + (size_t)img * IMGN + w;
    for (int i = t; i < 512; i += 256) { float2 z = base[(size_t)i * 512]; re[i] = z.x; im[i] = z.y; }
    __syncthreads();
    fft512<SIGN>(re, im, t);
    __syncthreads();
    for (int i = t; i < 512; i += 256) base[(size_t)i * 512] = make_float2(re[i], im[i]);
}

// ---------------------------------------------------------------------------
// Conv 3x3 (pad 1) as implicit GEMM on v_wmma_f32_16x16x32_f16.
// One wave (32 threads) computes 16 out-channels x (4x4 pixel patch).
// LDS tiles are staged in *fragment-ready* layout: each lane's v16h operand
// for K-step s is 32 contiguous, 32B-aligned bytes at frag[(s*32+lane)*16],
// so operand loads are 2x ds_load_b128 instead of 16x ds_load_u16.
// RELU_POOL: relu + 2x2 avgpool -> out (B, OCT*16, HW/2, HW/2)
// !RELU_POOL: spatial-sum atomicAdd  -> out (B, OCT*16)
// FUSED: CIN=6 input = concat(x, log10(|fftshift(spec)|+1)) computed on the fly.
// ---------------------------------------------------------------------------
template<int CIN, int OCT, int HW, bool FUSED, bool RELU_POOL>
__global__ __launch_bounds__(32)
void conv3x3_wmma(const float* __restrict__ in, const float2* __restrict__ spec,
                  const float* __restrict__ wgt, const float* __restrict__ bias,
                  float* __restrict__ out) {
    constexpr int K     = CIN * 9;
    constexpr int KP    = ((K + 31) / 32) * 32;
    constexpr int NSTEP = KP / 32;
    __shared__ __align__(32) _Float16 Afrag[NSTEP * 32 * 16];
    __shared__ __align__(32) _Float16 Bfrag[NSTEP * 32 * 16];
    __shared__ float Ct[16 * 16];

    const int lane = threadIdx.x;
    const int b   = blockIdx.z / OCT;
    const int oc0 = (blockIdx.z % OCT) * 16;
    const int h0  = blockIdx.y * 4;
    const int w0  = blockIdx.x * 4;

    // stage A: weights (COUT, CIN, 3, 3) contiguous; scatter into A-fragment
    // layout.  A element at (row m, col k):
    //   kr=k%32: kr<16 -> half=kr>>3, j=(kr&7)>>1 ; kr>=16 -> half=(kr>>3)&1,
    //   j=4+((kr&7)>>1) ; t=kr&1 ; dest lane = half*16+m, elem = 2j+t.
    for (int i = lane; i < 16 * KP; i += 32) {
        int m = i / KP, k = i % KP;
        float v = 0.0f;
        if (k < K) v = wgt[(size_t)(oc0 + m) * K + k];
        int step = k >> 5, kr = k & 31;
        int half, j;
        if (kr < 16) { half = kr >> 3;       j = (kr & 7) >> 1; }
        else         { half = (kr >> 3) & 1; j = 4 + ((kr & 7) >> 1); }
        int t = kr & 1;
        Afrag[((step * 32) + (half * 16 + m)) * 16 + 2 * j + t] = (_Float16)v;
    }
    // stage B: im2col patch, n = (dy)*4 + dx over a 4x4 spatial tile.
    // B element at (row k, col n): kr=k%32 -> half=(kr>>3)&1,
    // j=((kr&7)>>1)+4*(kr>>4), t=kr&1 ; dest lane = half*16+n, elem = 2j+t.
    for (int i = lane; i < KP * 16; i += 32) {
        int k = i / 16, n = i % 16;
        float v = 0.0f;
        if (k < K) {
            int c  = k / 9;
            int kh = (k % 9) / 3;
            int kw = k % 3;
            int ph = h0 + (n >> 2) + kh - 1;
            int pw = w0 + (n & 3)  + kw - 1;
            if (ph >= 0 && ph < HW && pw >= 0 && pw < HW) {
                if constexpr (FUSED) {
                    if (c < 3) {
                        v = in[(((size_t)b * 3 + c) * 512 + ph) * 512 + pw];
                    } else {
                        int c2 = (c - 3 + 2) % 3;        // channel fftshift (n=3)
                        int bb = (b + 4) & 7;            // batch fftshift (n=8)
                        int hh = (ph + 256) & 511;
                        int ww = (pw + 256) & 511;
                        float2 z = spec[(((size_t)bb * 3 + c2) * 512 + hh) * 512 + ww];
                        v = log10f(sqrtf(z.x * z.x + z.y * z.y) + 1.0f);
                    }
                } else {
                    v = in[(((size_t)b * CIN + c) * HW + ph) * HW + pw];
                }
            }
        }
        int step = k >> 5, kr = k & 31;
        int half = (kr >> 3) & 1, j = ((kr & 7) >> 1) + 4 * (kr >> 4), t = kr & 1;
        Bfrag[((step * 32) + (half * 16 + n)) * 16 + 2 * j + t] = (_Float16)v;
    }
    __syncthreads();

    const int half = lane >> 4;
    const int l16  = lane & 15;
    v8f acc = {};
#pragma unroll
    for (int s = 0; s < NSTEP; ++s) {
        v16h a  = *reinterpret_cast<const v16h*>(&Afrag[(s * 32 + lane) * 16]);
        v16h bf = *reinterpret_cast<const v16h*>(&Bfrag[(s * 32 + lane) * 16]);
        acc = __builtin_amdgcn_wmma_f32_16x16x32_f16(false, a, false, bf,
                                                     (short)0, acc, false, false);
    }

    // epilogue: bias (+relu) into LDS C tile
#pragma unroll
    for (int r = 0; r < 8; ++r) {
        int m = r + 8 * half;
        float v = acc[r] + bias[oc0 + m];
        if (RELU_POOL) v = fmaxf(v, 0.0f);
        Ct[m * 16 + l16] = v;
    }
    __syncthreads();

    if constexpr (RELU_POOL) {
        constexpr int HWo = HW / 2;
        for (int i = lane; i < 64; i += 32) {
            int m = i >> 2, p = i & 3, pr = p >> 1, pc = p & 1;
            float s = Ct[m * 16 + (2 * pr) * 4 + 2 * pc]
                    + Ct[m * 16 + (2 * pr) * 4 + 2 * pc + 1]
                    + Ct[m * 16 + (2 * pr + 1) * 4 + 2 * pc]
                    + Ct[m * 16 + (2 * pr + 1) * 4 + 2 * pc + 1];
            int oh = (h0 >> 1) + pr, ow = (w0 >> 1) + pc;
            out[(((size_t)b * (OCT * 16) + oc0 + m) * HWo + oh) * HWo + ow] = 0.25f * s;
        }
    } else {
        if (lane < 16) {
            float s = 0.0f;
            for (int n2 = 0; n2 < 16; ++n2) s += Ct[lane * 16 + n2];
            atomicAdd(&out[(size_t)b * (OCT * 16) + oc0 + lane], s);
        }
    }
}

// ---------------------------------------------------------------------------
__global__ void zero_buf(float* p, int n) {
    int i = blockIdx.x * blockDim.x + threadIdx.x;
    if (i < n) p[i] = 0.0f;
}

// FC: value_set[b,:] = leaky_relu((mean @ fw1.T + fb1) @ fw2.T + fb2, 0.01)
__global__ __launch_bounds__(256)
void fc_kernel(const float* __restrict__ y3sum,
               const float* __restrict__ fw1, const float* __restrict__ fb1,
               const float* __restrict__ fw2, const float* __restrict__ fb2,
               float* __restrict__ vs) {
    __shared__ float ym[64];
    __shared__ float h[256];
    int b = blockIdx.x, t = threadIdx.x;
    if (t < 64) ym[t] = y3sum[b * 64 + t] * (1.0f / 16384.0f);  // /(128*128)
    __syncthreads();
    float acc = fb1[t];
    for (int i = 0; i < 64; ++i) acc += ym[i] * fw1[t * 64 + i];
    h[t] = acc;
    __syncthreads();
    if (t < 100) {
        float a2 = fb2[t];
        for (int i = 0; i < 256; ++i) a2 += h[i] * fw2[t * 256 + i];
        vs[b * 100 + t] = (a2 < 0.0f) ? 0.01f * a2 : a2;
    }
}

// fq_mask[b,h,w] = value_set[b, clip(searchsorted(radius_set, dist), 0, 99)]
__global__ void mask_kernel(const float* __restrict__ vs, float* __restrict__ fq) {
    int idx = blockIdx.x * blockDim.x + threadIdx.x;
    if (idx >= FQ_N) return;
    int b = idx >> 18;
    int h = (idx >> 9) & 511;
    int w = idx & 511;
    float dh = (float)h - 256.0f, dw = (float)w - 256.0f;
    float dist = sqrtf(dh * dh + dw * dw);
    const float step = 3.620386719675123f;   // sqrt(512^2*2)/2 * 0.01
    int bin = (int)ceilf(dist / step) - 1;
    bin = bin < 0 ? 0 : (bin > 99 ? 99 : bin);
    fq[idx] = vs[b * 100 + bin];
}

// spectrum *= ifftshift(mask)/N^2 (batch fftshift roll folded in)
__global__ void apply_mask(float2* __restrict__ cb, const float* __restrict__ fq) {
    int idx = blockIdx.x * blockDim.x + threadIdx.x;
    if (idx >= LOW_N) return;
    int w = idx & 511;
    int h = (idx >> 9) & 511;
    int b = idx / (3 * IMGN);
    float m = fq[((((b + 4) & 7) << 9 | ((h + 256) & 511)) << 9) | ((w + 256) & 511)]
              * (1.0f / 262144.0f);
    float2 z = cb[idx];
    cb[idx] = make_float2(z.x * m, z.y * m);
}

__global__ void finalize_kernel(const float2* __restrict__ cb, float* __restrict__ low) {
    int idx = blockIdx.x * blockDim.x + threadIdx.x;
    if (idx >= LOW_N) return;
    float v = cb[idx].x;
    low[idx] = fminf(fmaxf(v, 0.0f), 1.0f);
}

// ---------------------------------------------------------------------------
extern "C" void kernel_launch(void* const* d_in, const int* in_sizes, int n_in,
                              void* d_out, int out_size, void* d_ws, size_t ws_size,
                              hipStream_t stream) {
    const float* x   = (const float*)d_in[0];
    const float* w1  = (const float*)d_in[1];
    const float* b1  = (const float*)d_in[2];
    const float* w2  = (const float*)d_in[3];
    const float* b2  = (const float*)d_in[4];
    const float* w3  = (const float*)d_in[5];
    const float* b3  = (const float*)d_in[6];
    const float* fw1 = (const float*)d_in[7];
    const float* fb1 = (const float*)d_in[8];
    const float* fw2 = (const float*)d_in[9];
    const float* fb2 = (const float*)d_in[10];

    float*  of   = (float*)d_out;
    float*  low  = of;                 // [0, LOW_N)
    float*  fq   = of + LOW_N;         // [LOW_N, LOW_N+FQ_N)
    float*  vs   = of + LOW_N + FQ_N;  // 800 floats

    float*  wsf   = (float*)d_ws;
    float2* cb    = (float2*)wsf;              // 12,582,912 floats (complex spectrum)
    float*  y1p   = wsf + 12582912;            // 8,388,608 floats  (8,16,256,256)
    float*  y2p   = wsf + 20971520;            // 4,194,304 floats  (8,32,128,128)
    float*  y3sum = wsf + 25165824;            // 512 floats        (8,64)

    // forward FFT (w then h)
    fft_rows_real<<<12288, 256, 0, stream>>>(x, cb);
    fft_cols_c<-1><<<12288, 256, 0, stream>>>(cb);

    // conv1 (fi = concat(x, log-mag of shifted spectrum)) + relu + pool
    conv3x3_wmma<6, 1, 512, true, true>
        <<<dim3(128, 128, 8), 32, 0, stream>>>(x, cb, w1, b1, y1p);
    // conv2 + relu + pool
    conv3x3_wmma<16, 2, 256, false, true>
        <<<dim3(64, 64, 16), 32, 0, stream>>>(y1p, nullptr, w2, b2, y2p);
    // conv3 -> spatial sum (mean in FC)
    zero_buf<<<2, 256, 0, stream>>>(y3sum, 512);
    conv3x3_wmma<32, 4, 128, false, false>
        <<<dim3(32, 32, 32), 32, 0, stream>>>(y2p, nullptr, w3, b3, y3sum);

    // FC -> value_set, radial mask -> fq_mask
    fc_kernel<<<8, 256, 0, stream>>>(y3sum, fw1, fb1, fw2, fb2, vs);
    mask_kernel<<<(FQ_N + 255) / 256, 256, 0, stream>>>(vs, fq);

    // masked inverse FFT -> lowpass
    apply_mask<<<(LOW_N + 255) / 256, 256, 0, stream>>>(cb, fq);
    fft_rows_c<1><<<12288, 256, 0, stream>>>(cb);
    fft_cols_c<1><<<12288, 256, 0, stream>>>(cb);
    finalize_kernel<<<(LOW_N + 255) / 256, 256, 0, stream>>>(cb, low);

    (void)in_sizes; (void)n_in; (void)out_size; (void)ws_size;
}
